// MixedAttention_40286793236789
// MI455X (gfx1250) — matrix operand order repaired and checked by reading, not compiled, not run on hardware
//
#include <hip/hip_runtime.h>

typedef __attribute__((ext_vector_type(16))) _Float16 v16h;
typedef __attribute__((ext_vector_type(8)))  _Float16 v8h;
typedef __attribute__((ext_vector_type(4)))  _Float16 v4h;
typedef __attribute__((ext_vector_type(2)))  _Float16 v2h;
typedef __attribute__((ext_vector_type(8)))  float    v8f;
typedef __attribute__((address_space(3)))    _Float16 lds_f16;

// ---- CDNA5 WMMA f16 (16x16x32, f32 accum) --------------------------------
__device__ __forceinline__ v8f wmma_f32(v16h a, v16h b, v8f c) {
    // (neg_a, A, neg_b, B, c_mod, C, reuse_a, reuse_b)
    return __builtin_amdgcn_wmma_f32_16x16x32_f16(false, a, false, b,
                                                  (short)0, c, false, false);
}

// CDNA5 ISA f16 layouts (wave32):
//   A 16x32 (MxK): row = lane&15; element e -> K = 16*(e>>3) + 8*(lane>>4) + (e&7)
//     => per lane: TWO contiguous 8-half runs at K = 8g and K = 16+8g (g = lane>>4)
//   B 32x16 (KxN): col = lane&15; element e -> K = 16*(e>>3) + 8*(lane>>4) + (e&7)
//     => per lane: TWO contiguous 8-half runs at K = 8g and K = 16+8g (needs [n][k] storage)
//   C/D 16x16 f32: col = lane&15, row = r + 8*(lane>>4)

__device__ __forceinline__ v16h frag_cat(const _Float16* p0, const _Float16* p1) {
    v8h lo = *(const v8h*)p0;           // b128
    v8h hi = *(const v8h*)p1;           // b128
    return __builtin_shufflevector(lo, hi, 0, 1, 2, 3, 4, 5, 6, 7,
                                           8, 9, 10, 11, 12, 13, 14, 15);
}
// A fragment from row-major row base (K origin at row_base[0])
__device__ __forceinline__ v16h load_a(const _Float16* row_base, int g) {
    return frag_cat(row_base + 8 * g, row_base + 16 + 8 * g);
}
// B fragment from transposed-store row base ([n][k], K origin at row_base[0])
__device__ __forceinline__ v16h load_b(const _Float16* row_base, int g) {
    return frag_cat(row_base + 8 * g, row_base + 16 + 8 * g);
}
// LDS byte offset of a __shared__ object (AS3 pointer value == LDS offset)
__device__ __forceinline__ unsigned lds_off(const _Float16* p) {
    return (unsigned)(unsigned long long)(lds_f16*)p;
}

// ===========================================================================
// Kernel 1: projection GEMM  out[b,h,s,d] = X[4096,1024] @ W[1024,512] + bias
// Block tile 32(M) x 128(N), 8 waves = 2x4 grid of 16x32 sub-tiles.
// A staged row-major (f32->f16, packed b64 stores); B staged TRANSPOSED so
// every fragment load is ds_load_b128.  HASW2 folds lv1+lv2 at compile time.
// ===========================================================================
template <bool HASW2>
__global__ __launch_bounds__(256) void proj_gemm(
    const float* __restrict__ X, const float* __restrict__ W,
    const float* __restrict__ W2, const float* __restrict__ bias,
    const float* __restrict__ bias2, _Float16* __restrict__ out) {
    __shared__ __attribute__((aligned(16))) _Float16 ldsA[32][40];    // [m][k]
    __shared__ __attribute__((aligned(16))) _Float16 ldsBT[128][40];  // [n][k]
    const int wid = threadIdx.x >> 5, lane = threadIdx.x & 31;
    const int g = lane >> 4, half = lane & 15;
    const int m0 = blockIdx.x * 32;
    const int n0 = blockIdx.y * 128;
    const int rw = (wid >> 2) << 4;
    const int cw = (wid & 3) << 5;
    v8f acc0 = {}, acc1 = {};
    for (int k0 = 0; k0 < 1024; k0 += 32) {
        __syncthreads();
        {   // A: 32x32 f32 -> f16, one float4 load + one b64 LDS store/thread
            int rr = threadIdx.x >> 3;
            int kk = (threadIdx.x & 7) << 2;
            float4 x = *(const float4*)&X[(size_t)(m0 + rr) * 1024 + k0 + kk];
            v4h hh;
            hh[0] = (_Float16)x.x; hh[1] = (_Float16)x.y;
            hh[2] = (_Float16)x.z; hh[3] = (_Float16)x.w;
            *(v4h*)&ldsA[rr][kk] = hh;
        }
#pragma unroll
        for (int j = 0; j < 4; ++j) {    // B: 32x128, transposed into ldsBT
            int c = threadIdx.x + (j << 8);
            int kk = c >> 5;
            int nn = (c & 31) << 2;
            float4 w = *(const float4*)&W[(size_t)(k0 + kk) * 512 + n0 + nn];
            if (HASW2) {
                float4 w2 = *(const float4*)&W2[(size_t)(k0 + kk) * 512 + n0 + nn];
                w.x += w2.x; w.y += w2.y; w.z += w2.z; w.w += w2.w;
            }
            ldsBT[nn + 0][kk] = (_Float16)w.x;
            ldsBT[nn + 1][kk] = (_Float16)w.y;
            ldsBT[nn + 2][kk] = (_Float16)w.z;
            ldsBT[nn + 3][kk] = (_Float16)w.w;
        }
        __syncthreads();
        v16h a  = load_a(&ldsA[rw + half][0], g);
        v16h b0 = load_b(&ldsBT[cw + half][0], g);
        v16h b1 = load_b(&ldsBT[cw + 16 + half][0], g);
        acc0 = wmma_f32(a, b0, acc0);
        acc1 = wmma_f32(a, b1, acc1);
    }
    float bv0 = bias[n0 + cw + half];
    float bv1 = bias[n0 + cw + 16 + half];
    if (HASW2) { bv0 += bias2[n0 + cw + half]; bv1 += bias2[n0 + cw + 16 + half]; }
#pragma unroll
    for (int r = 0; r < 8; ++r) {
        int m = m0 + rw + r + (g << 3);
        int bb = m >> 11, ss = m & 2047;
        int n = n0 + cw + half;
        out[(((size_t)bb * 8 + (n >> 6)) * 2048 + ss) * 64 + (n & 63)] =
            (_Float16)(acc0[r] + bv0);
        n += 16;
        out[(((size_t)bb * 8 + (n >> 6)) * 2048 + ss) * 64 + (n & 63)] =
            (_Float16)(acc1[r] + bv1);
    }
}

// ===========================================================================
// Kernel 2: M = lk1^T @ lk1 per local (b,h) -> [16][64][64] f16
// ===========================================================================
__global__ __launch_bounds__(256) void local_mm(
    const _Float16* __restrict__ lk1, _Float16* __restrict__ Mout) {
    __shared__ __attribute__((aligned(16))) _Float16 tileT[64][40];  // [d][k-sample]
    const int bh = blockIdx.x;
    const _Float16* src = lk1 + (size_t)bh * 2048 * 64;
    const int wid = threadIdx.x >> 5, lane = threadIdx.x & 31;
    const int g = lane >> 4, half = lane & 15;
    const int m0 = (wid >> 1) << 4;
    const int n0 = (wid & 1) << 5;
    v8f acc0 = {}, acc1 = {};
    for (int k0 = 0; k0 < 2048; k0 += 32) {
        __syncthreads();
        {   // stage 32 samples x 64 dims, transposed
            int kk = threadIdx.x >> 3;
            int c = (threadIdx.x & 7) << 3;
            v8h x = *(const v8h*)&src[(size_t)(k0 + kk) * 64 + c];
#pragma unroll
            for (int i = 0; i < 8; ++i) tileT[c + i][kk] = x[i];
        }
        __syncthreads();
        v16h a  = load_a(&tileT[m0 + half][0], g);       // A[m,k]=lk1[k,m]
        v16h b0 = load_b(&tileT[n0 + half][0], g);       // B[k,n]=lk1[k,n]
        v16h b1 = load_b(&tileT[n0 + 16 + half][0], g);
        acc0 = wmma_f32(a, b0, acc0);
        acc1 = wmma_f32(a, b1, acc1);
    }
    _Float16* Md = Mout + (size_t)bh * 64 * 64;
#pragma unroll
    for (int r = 0; r < 8; ++r) {
        int m = m0 + r + (g << 3);
        Md[m * 64 + n0 + half]      = (_Float16)acc0[r];
        Md[m * 64 + n0 + 16 + half] = (_Float16)acc1[r];
    }
}

// ===========================================================================
// Kernel 3: Q' = lq @ M ([2048,64] x [64,64]) per local (b,h)
// ===========================================================================
__global__ __launch_bounds__(256) void qprime_gemm(
    const _Float16* __restrict__ lq, const _Float16* __restrict__ Mw,
    _Float16* __restrict__ qp) {
    __shared__ __attribute__((aligned(16))) _Float16 mtT[64][72];  // [n][k]
    const int bh = blockIdx.y;
    const _Float16* Q  = lq + (size_t)bh * 2048 * 64;
    const _Float16* Md = Mw + (size_t)bh * 64 * 64;
    _Float16* O = qp + (size_t)bh * 2048 * 64;
    const int wid = threadIdx.x >> 5, lane = threadIdx.x & 31;
    const int g = lane >> 4, half = lane & 15;
    const int i0 = blockIdx.x * 128 + (wid << 4);
#pragma unroll
    for (int j = 0; j < 2; ++j) {        // cache 64x64 M transposed
        int c = threadIdx.x + (j << 8);
        int row = c >> 3, cc = (c & 7) << 3;
        v8h x = *(const v8h*)&Md[row * 64 + cc];
#pragma unroll
        for (int i = 0; i < 8; ++i) mtT[cc + i][row] = x[i];
    }
    __syncthreads();
    v8f acc0 = {}, acc1 = {}, acc2 = {}, acc3 = {};
#pragma unroll
    for (int k0 = 0; k0 < 64; k0 += 32) {
        v16h a = load_a(&Q[(size_t)(i0 + half) * 64 + k0], g);   // global b128 x2
        acc0 = wmma_f32(a, load_b(&mtT[0  + half][k0], g), acc0);
        acc1 = wmma_f32(a, load_b(&mtT[16 + half][k0], g), acc1);
        acc2 = wmma_f32(a, load_b(&mtT[32 + half][k0], g), acc2);
        acc3 = wmma_f32(a, load_b(&mtT[48 + half][k0], g), acc3);
    }
#pragma unroll
    for (int r = 0; r < 8; ++r) {
        int s = i0 + r + (g << 3);
        O[(size_t)s * 64 + 0  + half] = (_Float16)acc0[r];
        O[(size_t)s * 64 + 16 + half] = (_Float16)acc1[r];
        O[(size_t)s * 64 + 32 + half] = (_Float16)acc2[r];
        O[(size_t)s * 64 + 48 + half] = (_Float16)acc3[r];
    }
}

// ===========================================================================
// Kernel 4: unified flash attention.  K tile staged with CDNA5 async
// load-to-LDS DMA (GLOBAL_LOAD_ASYNC_TO_LDS_B128, ASYNCcnt); V staged
// transposed with key columns interleaved (consistent permutation of the 32
// keys, P.V invariant); P scratch written as packed b32.  Online softmax
// with width-16 shfl_xor matching the C/D layout.
// ===========================================================================
__global__ __launch_bounds__(256) void attn(
    const _Float16* __restrict__ qg, const _Float16* __restrict__ kg,
    const _Float16* __restrict__ vg, const _Float16* __restrict__ qp,
    const _Float16* __restrict__ lk2, const _Float16* __restrict__ lvc,
    const float* __restrict__ mask, float* __restrict__ out) {
    __shared__ __attribute__((aligned(16))) _Float16 kt[32][72];     // [key][d]
    __shared__ __attribute__((aligned(16))) _Float16 vtT[64][40];    // [d][perm key]
    __shared__ __attribute__((aligned(16))) _Float16 scr[8][16 * 32];
    const int wid = threadIdx.x >> 5, lane = threadIdx.x & 31;
    const int g = lane >> 4, half = lane & 15;
    const int combo = blockIdx.y;        // b*16 + h
    const int b = combo >> 4;
    const int h = combo & 15;
    const bool is_global = (h < 8);
    const int bh = b * 8 + (h & 7);
    const _Float16* Q = (is_global ? qg : qp)  + (size_t)bh * 2048 * 64;
    const _Float16* K = (is_global ? kg : lk2) + (size_t)bh * 2048 * 64;
    const _Float16* V = (is_global ? vg : lvc) + (size_t)bh * 2048 * 64;
    const int i0 = blockIdx.x * 128 + (wid << 4);
    _Float16* my = scr[wid];
    const float scale = 0.125f;          // 1/sqrt(64)

    // per-thread staging coordinates
    const int skk = threadIdx.x >> 3;            // key row 0..31
    const int sc  = (threadIdx.x & 7) << 3;      // dim chunk
    const int skp = (skk < 16) ? (skk << 1) : (((skk - 16) << 1) | 1); // perm col
    const unsigned kt_dst = lds_off(&kt[skk][sc]);

    const v16h qa0 = load_a(&Q[(size_t)(i0 + half) * 64], g);        // dims 0..31
    const v16h qa1 = load_a(&Q[(size_t)(i0 + half) * 64 + 32], g);   // dims 32..63
    v8f acc0 = {}, acc1 = {}, acc2 = {}, acc3 = {};
    float mrow[8], lrow[8];
#pragma unroll
    for (int r = 0; r < 8; ++r) { mrow[r] = -3.0e38f; lrow[r] = 0.0f; }

    for (int j0 = 0; j0 < 2048; j0 += 32) {
        __syncthreads();
        {   // K: async DMA memory -> LDS, no VGPR roundtrip (ASYNCcnt)
            const _Float16* kgp = &K[(size_t)(j0 + skk) * 64 + sc];
            asm volatile("global_load_async_to_lds_b128 %0, %1, off"
                         :: "v"(kt_dst), "v"(kgp) : "memory");
            // V: b128 load + transposed scatter with interleaved key columns
            v8h xv = *(const v8h*)&V[(size_t)(j0 + skk) * 64 + sc];
#pragma unroll
            for (int i = 0; i < 8; ++i) vtT[sc + i][skp] = xv[i];
        }
        if (j0 + 32 < 2048)              // prefetch next V tile into caches
            __builtin_prefetch(&V[(size_t)(j0 + 32) * 64 + threadIdx.x * 8], 0, 0);
        asm volatile("s_wait_asynccnt 0x0" ::: "memory");
        __syncthreads();
        // ---- scores: B = K^T, e walks head dim (contiguous in kt rows) ----
        v8f s0 = {}, s1 = {};
        s0 = wmma_f32(qa0, load_b(&kt[half][0], g), s0);
        s0 = wmma_f32(qa1, load_b(&kt[half][32], g), s0);
        s1 = wmma_f32(qa0, load_b(&kt[16 + half][0], g), s1);
        s1 = wmma_f32(qa1, load_b(&kt[16 + half][32], g), s1);

        float madd0 = 0.0f, madd1 = 0.0f;
        if (is_global) {
            madd0 = mask[b * 2048 + j0 + half];
            madd1 = mask[b * 2048 + j0 + 16 + half];
        }
        // ---- online softmax over these 32 keys ----
#pragma unroll
        for (int r = 0; r < 8; ++r) {
            float x0 = s0[r] * scale + madd0;
            float x1 = s1[r] * scale + madd1;
            float t = fmaxf(x0, x1);
            t = fmaxf(t, __shfl_xor(t, 1, 16));
            t = fmaxf(t, __shfl_xor(t, 2, 16));
            t = fmaxf(t, __shfl_xor(t, 4, 16));
            t = fmaxf(t, __shfl_xor(t, 8, 16));
            float mn = fmaxf(mrow[r], t);
            float rescale = __expf(mrow[r] - mn);
            mrow[r] = mn;
            float p0 = __expf(x0 - mn), p1 = __expf(x1 - mn);
            float rs = p0 + p1;
            rs += __shfl_xor(rs, 1, 16);
            rs += __shfl_xor(rs, 2, 16);
            rs += __shfl_xor(rs, 4, 16);
            rs += __shfl_xor(rs, 8, 16);
            lrow[r] = lrow[r] * rescale + rs;
            int rr = r + (g << 3);                        // D-layout row
            v2h pp; pp[0] = (_Float16)p0; pp[1] = (_Float16)p1;
            *(v2h*)&my[rr * 32 + (half << 1)] = pp;       // packed b32 store
            acc0[r] *= rescale; acc1[r] *= rescale;
            acc2[r] *= rescale; acc3[r] *= rescale;
        }
        // ---- ctx += P @ V (P scratch read back as A: 2 b128) ----
        v16h pa = load_a(&my[half * 32], g);
        acc0 = wmma_f32(pa, load_b(&vtT[0  + half][0], g), acc0);
        acc1 = wmma_f32(pa, load_b(&vtT[16 + half][0], g), acc1);
        acc2 = wmma_f32(pa, load_b(&vtT[32 + half][0], g), acc2);
        acc3 = wmma_f32(pa, load_b(&vtT[48 + half][0], g), acc3);
    }
    // ---- epilogue: normalize, scatter to [B,S,16*64] f32 ----
#pragma unroll
    for (int r = 0; r < 8; ++r) {
        int s = i0 + r + (g << 3);
        float inv = 1.0f / lrow[r];
        float* o = out + ((size_t)(b * 2048 + s)) * 1024 + h * 64;
        o[0  + half] = acc0[r] * inv;
        o[16 + half] = acc1[r] * inv;
        o[32 + half] = acc2[r] * inv;
        o[48 + half] = acc3[r] * inv;
    }
}

// ===========================================================================
extern "C" void kernel_launch(void* const* d_in, const int* in_sizes, int n_in,
                              void* d_out, int out_size, void* d_ws, size_t ws_size,
                              hipStream_t stream) {
    (void)in_sizes; (void)n_in; (void)out_size; (void)ws_size;
    const float* X    = (const float*)d_in[0];
    const float* mask = (const float*)d_in[1];
    const float* wq   = (const float*)d_in[2];
    const float* bq   = (const float*)d_in[3];
    const float* wk   = (const float*)d_in[4];
    const float* bk   = (const float*)d_in[5];
    const float* wv   = (const float*)d_in[6];
    const float* bv   = (const float*)d_in[7];
    const float* wlq  = (const float*)d_in[8];
    const float* blq  = (const float*)d_in[9];
    const float* wlk1 = (const float*)d_in[10];
    const float* blk1 = (const float*)d_in[11];
    const float* wlk2 = (const float*)d_in[12];
    const float* blk2 = (const float*)d_in[13];
    const float* wlv1 = (const float*)d_in[14];
    const float* blv1 = (const float*)d_in[15];
    const float* wlv2 = (const float*)d_in[16];
    const float* blv2 = (const float*)d_in[17];

    _Float16* ws = (_Float16*)d_ws;
    const size_t P = (size_t)2 * 8 * 2048 * 64;   // one projection, [B,h,S,d]
    _Float16* q    = ws + 0 * P;
    _Float16* k    = ws + 1 * P;
    _Float16* v    = ws + 2 * P;
    _Float16* lqb  = ws + 3 * P;
    _Float16* lk1b = ws + 4 * P;
    _Float16* lk2b = ws + 5 * P;
    _Float16* lvc  = ws + 6 * P;                  // lv1 + lv2 (folded GEMM)
    _Float16* Mw   = ws + 7 * P;                  // 16 x 64 x 64
    _Float16* qp   = ws + 7 * P + (size_t)16 * 64 * 64;

    dim3 tb(256);
    dim3 pg(128, 4);                              // 4096/32 x 512/128
    proj_gemm<false><<<pg, tb, 0, stream>>>(X, wq,   nullptr, bq,   nullptr, q);
    proj_gemm<false><<<pg, tb, 0, stream>>>(X, wk,   nullptr, bk,   nullptr, k);
    proj_gemm<false><<<pg, tb, 0, stream>>>(X, wv,   nullptr, bv,   nullptr, v);
    proj_gemm<false><<<pg, tb, 0, stream>>>(X, wlq,  nullptr, blq,  nullptr, lqb);
    proj_gemm<false><<<pg, tb, 0, stream>>>(X, wlk1, nullptr, blk1, nullptr, lk1b);
    proj_gemm<false><<<pg, tb, 0, stream>>>(X, wlk2, nullptr, blk2, nullptr, lk2b);
    proj_gemm<true><<<pg, tb, 0, stream>>>(X, wlv1,  wlv2,    blv1, blv2,    lvc);

    local_mm<<<dim3(16), tb, 0, stream>>>(lk1b, Mw);
    qprime_gemm<<<dim3(16, 16), tb, 0, stream>>>(lqb, Mw, qp);

    attn<<<dim3(16, 32), tb, 0, stream>>>(q, k, v, qp, lk2b, lvc, mask,
                                          (float*)d_out);
}
